// AttnSleep_Improved_81973745811566
// MI455X (gfx1250) — matrix operand order repaired and checked by reading, not compiled, not run on hardware
//
#include <hip/hip_runtime.h>

// ======================= types & device helpers =======================
typedef __attribute__((ext_vector_type(16))) _Float16 v16h;
typedef __attribute__((ext_vector_type(8)))  float    v8f;

__device__ __forceinline__ float dev_gelu(float x) {
    return 0.5f * x * (1.0f + erff(x * 0.70710678118654752440f));
}
__device__ __forceinline__ float dev_sigmoid(float x) {
    return 1.0f / (1.0f + expf(-x));
}

__global__ void zero_kernel(float* __restrict__ p, int n)
{
    int i = blockIdx.x * blockDim.x + threadIdx.x;
    if (i < n) p[i] = 0.f;
}

// ======================= WMMA GEMM with fused epilogue =======================
// C[m,n] = act( rs[m]*cs[n] * (A@B + rb0[m] + cb0[n]) + rb1[m] + cb1[n] ) + addC
// act: 0=none 1=relu 2=gelu.  A: MxK (row-major, or KxM if transA), B: KxN row-major.
// REQUIRES K % 32 == 0 (all call sites use K in {64,128,256,512,6400}).
// One wave computes one 16x16 tile via v_wmma_f32_16x16x32_f16; batched over blockIdx.y.
//
// Operand layouts (wave32, 16x16x32 f16):
//   A  v16h: av[j] holds K = (j<8 ? j : j+8) + 8*(lane/16), M = lane%16
//   B  v16h: bv[j] holds K = (j<8 ? j : j+8) + 8*(lane/16), N = lane%16
//   C  v8f : acc[v] holds M = v + 8*(lane/16),              N = lane%16
//
// Edge tiles: out-of-range lanes are redirected to 'zbuf' (64 zeroed floats) with a
// ZERO stride, so every load is unconditional and in-bounds -> straight-line codegen.
__global__ void gemm_wmma_ep(const float* __restrict__ A, const float* __restrict__ Bm,
                             float* __restrict__ C, const float* __restrict__ zbuf,
                             int M, int N, int K, int lda, int ldb, int ldc,
                             long long sA, long long sB, long long sC,
                             int transA,
                             const float* __restrict__ rs, const float* __restrict__ rb0,
                             const float* __restrict__ rb1,
                             const float* __restrict__ cs, const float* __restrict__ cb0,
                             const float* __restrict__ cb1,
                             const float* __restrict__ addC, long long sAdd,
                             int act)
{
    const int wave = threadIdx.x >> 5;
    const int lane = threadIdx.x & 31;
    const int wavesPerBlock = blockDim.x >> 5;
    const int tilesN = (N + 15) >> 4;
    const int tilesM = (M + 15) >> 4;
    const int nTiles = tilesM * tilesN;
    int tileId = blockIdx.x * wavesPerBlock + wave;
    if (tileId >= nTiles) return;
    const int tm = tileId / tilesN;
    const int tn = tileId - tm * tilesN;
    const int batch = blockIdx.y;
    const float* Ab = A + (long long)batch * sA;
    const float* Bb = Bm + (long long)batch * sB;
    float* Cb = C + (long long)batch * sC;

    const int m0 = tm << 4;
    const int n0 = tn << 4;
    const int hi = lane >> 4;   // lane half selects K-subblock
    const int lm = lane & 15;   // M index (A) / N index (B, C)

    v8f acc = {0.f, 0.f, 0.f, 0.f, 0.f, 0.f, 0.f, 0.f};

    const bool interior = (m0 + 16 <= M) && (n0 + 16 <= N) && ((K & 63) == 0) && !transA;

    if (interior) {
        // -------- fast path: no guards, vectorized A loads, 2 WMMAs per iter --------
        const int mm = m0 + lm;
        const int nn = n0 + lm;
        const float* aRow = Ab + (long long)mm * lda + 8 * hi;
        const float* bCol = Bb + nn + (long long)(8 * hi) * ldb;
        for (int k0 = 0; k0 < K; k0 += 64) {
            v16h av0, bv0, av1, bv1;
            {
                const float* ap = aRow + k0;
                float4 p0 = *(const float4*)(ap);
                float4 p1 = *(const float4*)(ap + 4);
                float4 p2 = *(const float4*)(ap + 16);
                float4 p3 = *(const float4*)(ap + 20);
                float4 p4 = *(const float4*)(ap + 32);
                float4 p5 = *(const float4*)(ap + 36);
                float4 p6 = *(const float4*)(ap + 48);
                float4 p7 = *(const float4*)(ap + 52);
                av0[0]=(_Float16)p0.x; av0[1]=(_Float16)p0.y; av0[2]=(_Float16)p0.z; av0[3]=(_Float16)p0.w;
                av0[4]=(_Float16)p1.x; av0[5]=(_Float16)p1.y; av0[6]=(_Float16)p1.z; av0[7]=(_Float16)p1.w;
                av0[8]=(_Float16)p2.x; av0[9]=(_Float16)p2.y; av0[10]=(_Float16)p2.z; av0[11]=(_Float16)p2.w;
                av0[12]=(_Float16)p3.x; av0[13]=(_Float16)p3.y; av0[14]=(_Float16)p3.z; av0[15]=(_Float16)p3.w;
                av1[0]=(_Float16)p4.x; av1[1]=(_Float16)p4.y; av1[2]=(_Float16)p4.z; av1[3]=(_Float16)p4.w;
                av1[4]=(_Float16)p5.x; av1[5]=(_Float16)p5.y; av1[6]=(_Float16)p5.z; av1[7]=(_Float16)p5.w;
                av1[8]=(_Float16)p6.x; av1[9]=(_Float16)p6.y; av1[10]=(_Float16)p6.z; av1[11]=(_Float16)p6.w;
                av1[12]=(_Float16)p7.x; av1[13]=(_Float16)p7.y; av1[14]=(_Float16)p7.z; av1[15]=(_Float16)p7.w;
            }
            {
                const float* bp = bCol + (long long)k0 * ldb;
#pragma unroll
                for (int j = 0; j < 16; ++j)
                    bv0[j] = (_Float16)bp[(long long)(j < 8 ? j : j + 8) * ldb];
                const float* bp1 = bp + (long long)32 * ldb;
#pragma unroll
                for (int j = 0; j < 16; ++j)
                    bv1[j] = (_Float16)bp1[(long long)(j < 8 ? j : j + 8) * ldb];
            }
            acc = __builtin_amdgcn_wmma_f32_16x16x32_f16(false, av0, false, bv0,
                                                         (short)0, acc, false, false);
            acc = __builtin_amdgcn_wmma_f32_16x16x32_f16(false, av1, false, bv1,
                                                         (short)0, acc, false, false);
        }
    } else {
        // -------- edge path: zero-stride redirect, fully branch-free loads --------
        const int mreal = m0 + lm;
        const int nreal = n0 + lm;
        const bool mok = mreal < M;
        const bool nok = nreal < N;
        const long long asM = transA ? 1 : lda;   // stride per M step
        const long long asK = transA ? lda : 1;   // stride per K step
        const float* aP = mok ? (Ab + (long long)mreal * asM) : zbuf;
        const long long aStep = mok ? asK : 0;
        const float* bP = nok ? (Bb + nreal) : zbuf;
        const long long bStep = nok ? (long long)ldb : 0;
        for (int k0 = 0; k0 < K; k0 += 32) {
            v16h av, bv;
#pragma unroll
            for (int j = 0; j < 16; ++j) {
                int ka = k0 + (j < 8 ? j : j + 8) + 8 * hi;
                av[j] = (_Float16)aP[(long long)ka * aStep];
                int kb = k0 + (j < 8 ? j : j + 8) + 8 * hi;
                bv[j] = (_Float16)bP[(long long)kb * bStep];
            }
            acc = __builtin_amdgcn_wmma_f32_16x16x32_f16(false, av, false, bv,
                                                         (short)0, acc, false, false);
        }
    }

#pragma unroll
    for (int vv = 0; vv < 8; ++vv) {
        int mm = m0 + vv + 8 * hi;     // C/D f32 layout: M = v + 8*hi
        int nn = n0 + lm;
        if (mm < M && nn < N) {
            float xv = acc[vv];
            float pre = xv + (rb0 ? rb0[mm] : 0.f) + (cb0 ? cb0[nn] : 0.f);
            float scl = (rs ? rs[mm] : 1.f) * (cs ? cs[nn] : 1.f);
            float y = scl * pre + (rb1 ? rb1[mm] : 0.f) + (cb1 ? cb1[nn] : 0.f);
            if (act == 1) y = fmaxf(y, 0.f);
            else if (act == 2) y = dev_gelu(y);
            long long oidx = (long long)mm * ldc + nn;
            if (addC) y += addC[(long long)batch * sAdd + oidx];
            Cb[oidx] = y;
        }
    }
}

// ======================= elementwise / conv kernels =======================
__global__ void dwconv_kernel(const float* __restrict__ in, const float* __restrict__ w,
                              const float* __restrict__ bias, float* __restrict__ out,
                              int B, int C, int Lin, int Lout, int k, int stride, int pad)
{
    long long idx = (long long)blockIdx.x * blockDim.x + threadIdx.x;
    long long total = (long long)B * C * Lout;
    if (idx >= total) return;
    int lo = (int)(idx % Lout);
    int c  = (int)((idx / Lout) % C);
    long long b = idx / ((long long)Lout * C);
    const float* ip = in + ((long long)b * C + c) * Lin;
    const float* wp = w + (long long)c * k;
    int s0 = lo * stride - pad;
    float acc = bias[c];
    for (int j = 0; j < k; ++j) {
        int l = s0 + j;
        if (l >= 0 && l < Lin) acc += wp[j] * ip[l];
    }
    out[idx] = acc;
}

// pointwise 1->64 + BN + gelu (Cin==1 case where a WMMA GEMM would be pure padding)
__global__ void pw_c1_kernel(const float* __restrict__ x, const float* __restrict__ pw_w,
                             const float* __restrict__ pw_b, const float* __restrict__ g,
                             const float* __restrict__ bb, float* __restrict__ out,
                             int B, int L)
{
    long long idx = (long long)blockIdx.x * blockDim.x + threadIdx.x;
    long long total = (long long)B * 64 * L;
    if (idx >= total) return;
    int l  = (int)(idx % L);
    int co = (int)((idx / L) % 64);
    long long b = idx / ((long long)L * 64);
    float y = g[co] * (pw_w[co] * x[b * L + l] + pw_b[co]) + bb[co];
    out[idx] = dev_gelu(y);
}

__global__ void maxpool_kernel(const float* __restrict__ in, float* __restrict__ out,
                               int B, int C, int Lin, int Lout, int k, int s, int p)
{
    long long idx = (long long)blockIdx.x * blockDim.x + threadIdx.x;
    long long total = (long long)B * C * Lout;
    if (idx >= total) return;
    int lo = (int)(idx % Lout);
    int c  = (int)((idx / Lout) % C);
    long long b = idx / ((long long)Lout * C);
    const float* ip = in + ((long long)b * C + c) * Lin;
    int s0 = lo * s - p;
    float m = -3.4e38f;
    for (int j = 0; j < k; ++j) {
        int l = s0 + j;
        if (l >= 0 && l < Lin) m = fmaxf(m, ip[l]);
    }
    out[idx] = m;
}

__global__ void ca_stats_kernel(const float* __restrict__ x, float* __restrict__ avg,
                                float* __restrict__ mx, int L)
{
    int bc = blockIdx.x;
    const float* p = x + (long long)bc * L;
    float s = 0.f, m = -3.4e38f;
    for (int i = threadIdx.x; i < L; i += blockDim.x) {
        float v = p[i]; s += v; m = fmaxf(m, v);
    }
    __shared__ float ss[128], sm[128];
    ss[threadIdx.x] = s; sm[threadIdx.x] = m;
    __syncthreads();
    for (int o = 64; o > 0; o >>= 1) {
        if ((int)threadIdx.x < o) {
            ss[threadIdx.x] += ss[threadIdx.x + o];
            sm[threadIdx.x] = fmaxf(sm[threadIdx.x], sm[threadIdx.x + o]);
        }
        __syncthreads();
    }
    if (threadIdx.x == 0) { avg[bc] = ss[0] / (float)L; mx[bc] = sm[0]; }
}

// channel-attention MLP (C=64, hidden=4): att = sigmoid(mlp(avg)+mlp(max))
__global__ void ca_apply_kernel(const float* __restrict__ avg, const float* __restrict__ mx,
                                const float* __restrict__ fc1w, const float* __restrict__ fc1b,
                                const float* __restrict__ fc2w, const float* __restrict__ fc2b,
                                float* __restrict__ att)
{
    int b = blockIdx.x; int c = threadIdx.x;
    __shared__ float sav[64], smx[64], hav[4], hmx[4];
    sav[c] = avg[b * 64 + c]; smx[c] = mx[b * 64 + c];
    __syncthreads();
    if (c < 8) {
        int j = c & 3;
        const float* src = (c < 4) ? sav : smx;
        float acc = fc1b[j];
        for (int i = 0; i < 64; ++i) acc += src[i] * fc1w[i * 4 + j];
        acc = fmaxf(acc, 0.f);
        if (c < 4) hav[j] = acc; else hmx[j] = acc;
    }
    __syncthreads();
    float s = 2.f * fc2b[c];
    for (int j = 0; j < 4; ++j) s += (hav[j] + hmx[j]) * fc2w[j * 64 + c];
    att[b * 64 + c] = dev_sigmoid(s);
}

__global__ void scale_rows_kernel(float* __restrict__ x, const float* __restrict__ att,
                                  int L, long long total)
{
    long long idx = (long long)blockIdx.x * blockDim.x + threadIdx.x;
    if (idx >= total) return;
    x[idx] *= att[idx / L];
}

__global__ void sa_stats_kernel(const float* __restrict__ x, float* __restrict__ sp,
                                int B, int C, int L)
{
    long long idx = (long long)blockIdx.x * blockDim.x + threadIdx.x;
    long long total = (long long)B * L;
    if (idx >= total) return;
    int l = (int)(idx % L);
    long long b = idx / L;
    const float* p = x + (long long)b * C * L + l;
    float s = 0.f, m = -3.4e38f;
    for (int c = 0; c < C; ++c) { float v = p[(long long)c * L]; s += v; m = fmaxf(m, v); }
    sp[(b * 2) * L + l] = s / (float)C;
    sp[(b * 2 + 1) * L + l] = m;
}

__global__ void sa_scale_kernel(float* __restrict__ x, const float* __restrict__ sp,
                                const float* __restrict__ w, const float* __restrict__ bb,
                                int B, int C, int L)
{
    long long idx = (long long)blockIdx.x * blockDim.x + threadIdx.x;
    long long total = (long long)B * L;
    if (idx >= total) return;
    int l = (int)(idx % L);
    long long b = idx / L;
    const float* mean = sp + b * 2 * L;
    const float* mxv  = mean + L;
    float acc = bb[0];
    for (int j = 0; j < 7; ++j) {
        int ll = l - 3 + j;
        if (ll >= 0 && ll < L) acc += w[j] * mean[ll] + w[7 + j] * mxv[ll];
    }
    float sig = dev_sigmoid(acc);
    float* p = x + (long long)b * C * L + l;
    for (int c = 0; c < C; ++c) p[(long long)c * L] *= sig;
}

// torch-semantics adaptive average pool; writes into [B,Ctot,O] at channel offset coff
__global__ void adaptive_kernel(const float* __restrict__ in, float* __restrict__ out,
                                int B, int C, int Lin, int O, int Ctot, int coff)
{
    long long idx = (long long)blockIdx.x * blockDim.x + threadIdx.x;
    long long total = (long long)B * C * O;
    if (idx >= total) return;
    int i = (int)(idx % O);
    int c = (int)((idx / O) % C);
    long long b = idx / ((long long)O * C);
    int s = (int)(((long long)i * Lin) / O);
    int e = (int)((((long long)(i + 1)) * Lin + O - 1) / O);
    const float* p = in + ((long long)b * C + c) * Lin;
    float acc = 0.f;
    for (int l = s; l < e; ++l) acc += p[l];
    out[((long long)b * Ctot + c + coff) * O + i] = acc / (float)(e - s);
}

// out[b, i2, i1] = in[b, i1, i2]  (in dims B x D1 x D2)
__global__ void transpose_b_kernel(const float* __restrict__ in, float* __restrict__ out,
                                   int B, int D1, int D2)
{
    long long idx = (long long)blockIdx.x * blockDim.x + threadIdx.x;
    long long total = (long long)B * D1 * D2;
    if (idx >= total) return;
    int i2 = (int)(idx % D2);
    int i1 = (int)((idx / D2) % D1);
    long long b = idx / ((long long)D1 * D2);
    out[((long long)b * D2 + i2) * D1 + i1] = in[idx];
}

// causal conv 64->64, k=7, left pad 6 on [B,64,100]
__global__ void causal_conv_kernel(const float* __restrict__ x, const float* __restrict__ w,
                                   const float* __restrict__ bias, float* __restrict__ out, int B)
{
    long long idx = (long long)blockIdx.x * blockDim.x + threadIdx.x;
    long long total = (long long)B * 64 * 100;
    if (idx >= total) return;
    int t  = (int)(idx % 100);
    int co = (int)((idx / 100) % 64);
    long long b = idx / 6400;
    const float* xb = x + b * 64 * 100;
    const float* wc = w + (long long)co * 64 * 7;
    float acc = bias[co];
    for (int ci = 0; ci < 64; ++ci) {
        const float* xc = xb + ci * 100;
        const float* wk = wc + ci * 7;
#pragma unroll
        for (int j = 0; j < 7; ++j) {
            int tt = t - 6 + j;
            if (tt >= 0) acc += wk[j] * xc[tt];
        }
    }
    out[idx] = acc;
}

// fused attention per (b,h): scores = QK^T*0.25 + rel_bias -> softmax -> @V
// q,k,v are [B,64,100] channels-first; out is [B*T, 64] tokens-major
__global__ void attention_kernel(const float* __restrict__ q, const float* __restrict__ k,
                                 const float* __restrict__ v, const float* __restrict__ relb,
                                 float* __restrict__ out)
{
    const int T = 100, H = 8, D = 8, C = 64;
    int bh = blockIdx.x;
    int b = bh / H, h = bh % H;
    __shared__ float qs[100][8], ks[100][8], vs[100][8];
    __shared__ float sc[100][101];
    long long base = ((long long)b * C + h * D) * T;
    for (int i = threadIdx.x; i < T * D; i += blockDim.x) {
        int t = i / D, d = i % D;
        qs[t][d] = q[base + (long long)d * T + t];
        ks[t][d] = k[base + (long long)d * T + t];
        vs[t][d] = v[base + (long long)d * T + t];
    }
    __syncthreads();
    int t = threadIdx.x;
    if (t < T) {
        const float* rb = relb + ((long long)h * T + t) * T;
        float mxv = -3.4e38f;
        for (int s = 0; s < T; ++s) {
            float dot = 0.f;
#pragma unroll
            for (int d = 0; d < D; ++d) dot += qs[t][d] * ks[s][d];
            float val = dot * 0.25f + rb[s];   // DK_SCALE = 1/sqrt(128/8)
            sc[t][s] = val;
            mxv = fmaxf(mxv, val);
        }
        float sum = 0.f;
        for (int s = 0; s < T; ++s) { float e = expf(sc[t][s] - mxv); sc[t][s] = e; sum += e; }
        float inv = 1.f / sum;
        float acc[8];
#pragma unroll
        for (int d = 0; d < D; ++d) acc[d] = 0.f;
        for (int s = 0; s < T; ++s) {
            float wgt = sc[t][s];
#pragma unroll
            for (int d = 0; d < D; ++d) acc[d] += wgt * vs[s][d];
        }
        for (int d = 0; d < D; ++d)
            out[((long long)(b * T + t)) * C + h * D + d] = acc[d] * inv;
    }
}

// x_tok[b,t,c] = q_cf[b,c,t] + proj[b*T+t, c]
__global__ void residual1_kernel(const float* __restrict__ qcf, const float* __restrict__ proj,
                                 float* __restrict__ xt, int B)
{
    long long idx = (long long)blockIdx.x * blockDim.x + threadIdx.x;
    long long total = (long long)B * 100 * 64;
    if (idx >= total) return;
    int c = (int)(idx % 64);
    int t = (int)((idx / 64) % 100);
    long long b = idx / 6400;
    xt[idx] = qcf[((long long)b * 64 + c) * 100 + t] + proj[idx];
}

__global__ void layernorm_kernel(const float* __restrict__ x, const float* __restrict__ g,
                                 const float* __restrict__ bb, float* __restrict__ out,
                                 int rows, int C)
{
    int r = blockIdx.x * blockDim.x + threadIdx.x;
    if (r >= rows) return;
    const float* p = x + (long long)r * C;
    float m = 0.f;
    for (int c = 0; c < C; ++c) m += p[c];
    m /= (float)C;
    float var = 0.f;
    for (int c = 0; c < C; ++c) { float d = p[c] - m; var += d * d; }
    var /= (float)C;
    float inv = rsqrtf(var + 1e-6f);
    float* o = out + (long long)r * C;
    for (int c = 0; c < C; ++c) o[c] = g[c] * (p[c] - m) * inv + bb[c];
}

// ======================= host-side helpers =======================
static inline unsigned gblk(long long n) { return (unsigned)((n + 255) / 256); }

static const float* g_zbuf = nullptr;  // set per launch before any GEMM

static void launch_gemm(hipStream_t st, const float* A, const float* Bm, float* C,
                        int M, int N, int K, int lda, int ldb, int ldc,
                        long long sA, long long sB, long long sC, int batch, int transA,
                        const float* rs, const float* rb0, const float* rb1,
                        const float* cs, const float* cb0, const float* cb1,
                        const float* addC, long long sAdd, int act)
{
    int tiles = ((M + 15) / 16) * ((N + 15) / 16);
    dim3 grid((unsigned)((tiles + 7) / 8), (unsigned)batch);
    gemm_wmma_ep<<<grid, 256, 0, st>>>(A, Bm, C, g_zbuf, M, N, K, lda, ldb, ldc,
                                       sA, sB, sC, transA,
                                       rs, rb0, rb1, cs, cb0, cb1, addC, sAdd, act);
}

// ======================= driver =======================
extern "C" void kernel_launch(void* const* d_in, const int* in_sizes, int n_in,
                              void* d_out, int out_size, void* d_ws, size_t ws_size,
                              hipStream_t stream)
{
    (void)in_sizes; (void)out_size; (void)ws_size;
    if (n_in < 156) return;  // expects jax-pytree (sorted-dict) flattening: x + 155 param leaves

    const int B = 64, T = 100, C = 64;
    const float* x = (const float*)d_in[0];
    auto P = [&](int i) -> const float* { return (const float*)d_in[i]; };

    char* wsb = (char*)d_ws;
    size_t off = 0;
    auto alloc = [&](size_t nFloats) -> float* {
        float* p = (float*)(wsb + off);
        off += ((nFloats * sizeof(float) + 255) & ~(size_t)255);
        return p;
    };

    // zero page for edge-tile redirect loads (ws is poisoned, so clear every launch)
    float* zbuf = alloc(64);
    zero_kernel<<<1, 64, 0, stream>>>(zbuf, 64);
    g_zbuf = zbuf;

    // dsconv (Cin>=16): depthwise kernel, then WMMA pointwise GEMM with fused BN+gelu
    auto run_dsconv = [&](const float* in, int Cin, int Cout, int Lin,
                          int kk, int stride, int pad, int pbase, int& Lout) -> float* {
        const float* bn_b = P(pbase + 0); const float* bn_g = P(pbase + 1);
        const float* dw_b = P(pbase + 2); const float* dw_w = P(pbase + 3);
        const float* pw_b = P(pbase + 4); const float* pw_w = P(pbase + 5);
        Lout = (Lin + 2 * pad - kk) / stride + 1;
        float* dwout = alloc((size_t)B * Cin * Lout);
        long long tot = (long long)B * Cin * Lout;
        dwconv_kernel<<<gblk(tot), 256, 0, stream>>>(in, dw_w, dw_b, dwout, B, Cin, Lin, Lout, kk, stride, pad);
        float* outb = alloc((size_t)B * Cout * Lout);
        launch_gemm(stream, pw_w, dwout, outb, Cout, Lout, Cin, Cin, Lout, Lout,
                    0, (long long)Cin * Lout, (long long)Cout * Lout, B, 0,
                    bn_g, pw_b, bn_b, nullptr, nullptr, nullptr, nullptr, 0, 2);
        return outb;
    };

    auto run_maxpool = [&](const float* in, int Cc, int Lin, int kk, int s, int p, int& Lout) -> float* {
        Lout = (Lin + 2 * p - kk) / s + 1;
        float* o = alloc((size_t)B * Cc * Lout);
        maxpool_kernel<<<gblk((long long)B * Cc * Lout), 256, 0, stream>>>(in, o, B, Cc, Lin, Lout, kk, s, p);
        return o;
    };

    auto run_cbam = [&](float* xb, int Cc, int L, int pbase) {
        float* avg = alloc((size_t)B * Cc);
        float* mxv = alloc((size_t)B * Cc);
        float* att = alloc((size_t)B * Cc);
        ca_stats_kernel<<<B * Cc, 128, 0, stream>>>(xb, avg, mxv, L);
        ca_apply_kernel<<<B, 64, 0, stream>>>(avg, mxv, P(pbase + 1), P(pbase + 0),
                                              P(pbase + 3), P(pbase + 2), att);
        long long tot = (long long)B * Cc * L;
        scale_rows_kernel<<<gblk(tot), 256, 0, stream>>>(xb, att, L, tot);
        float* sp = alloc((size_t)B * 2 * L);
        sa_stats_kernel<<<gblk((long long)B * L), 256, 0, stream>>>(xb, sp, B, Cc, L);
        sa_scale_kernel<<<gblk((long long)B * L), 256, 0, stream>>>(xb, sp, P(pbase + 5), P(pbase + 4), B, Cc, L);
    };

    float* xc = alloc((size_t)B * 256 * 100);  // fused MRCNN feature map [B,256,100]

    // ---------------- features1 ----------------
    {
        int L1 = (30000 + 48 - 50) / 6 + 1;                       // 5000
        float* dw1 = alloc((size_t)B * L1);
        dwconv_kernel<<<gblk((long long)B * L1), 256, 0, stream>>>(x, P(10), P(9), dw1, B, 1, 30000, L1, 50, 6, 24);
        float* x1 = alloc((size_t)B * 64 * L1);
        pw_c1_kernel<<<gblk((long long)B * 64 * L1), 256, 0, stream>>>(dw1, P(12), P(11), P(8), P(7), x1, B, L1);
        int L2; float* mp1 = run_maxpool(x1, 64, L1, 8, 2, 4, L2); // 2501
        run_cbam(mp1, 64, L2, 13);
        int L3; float* x2 = run_dsconv(mp1, 64, 128, L2, 8, 1, 4, 19, L3);   // 2502
        int L4; float* x3 = run_dsconv(x2, 128, 128, L3, 8, 1, 4, 25, L4);   // 2503
        int L5; float* mp2 = run_maxpool(x3, 128, L4, 4, 4, 2, L5);          // 626
        adaptive_kernel<<<gblk((long long)B * 128 * 100), 256, 0, stream>>>(mp2, xc, B, 128, L5, 100, 256, 0);
    }
    // ---------------- features2 ----------------
    {
        int M1 = (30000 + 400 - 400) / 50 + 1;                    // 601
        float* dw1 = alloc((size_t)B * M1);
        dwconv_kernel<<<gblk((long long)B * M1), 256, 0, stream>>>(x, P(34), P(33), dw1, B, 1, 30000, M1, 400, 50, 200);
        float* x1 = alloc((size_t)B * 64 * M1);
        pw_c1_kernel<<<gblk((long long)B * 64 * M1), 256, 0, stream>>>(dw1, P(36), P(35), P(32), P(31), x1, B, M1);
        int M2; float* mp1 = run_maxpool(x1, 64, M1, 4, 2, 2, M2); // 301
        run_cbam(mp1, 64, M2, 37);
        int M3; float* x2 = run_dsconv(mp1, 64, 128, M2, 7, 1, 3, 43, M3);   // 301
        int M4; float* x3 = run_dsconv(x2, 128, 128, M3, 7, 1, 3, 49, M4);   // 301
        int M5; float* mp2 = run_maxpool(x3, 128, M4, 2, 2, 1, M5);          // 151
        adaptive_kernel<<<gblk((long long)B * 128 * 100), 256, 0, stream>>>(mp2, xc, B, 128, M5, 100, 256, 128);
    }
    // ---------------- fusion ----------------
    int Lf;
    float* xf = run_dsconv(xc, 256, 64, 100, 3, 1, 1, 55, Lf);    // [B,64,100]
    run_cbam(xf, 64, 100, 61);
    float* feat = alloc((size_t)B * 64 * 100);
    adaptive_kernel<<<gblk((long long)B * 64 * 100), 256, 0, stream>>>(xf, feat, B, 64, 100, 100, 64, 0);

    // ---------------- transformer ----------------
    float* tokens  = alloc((size_t)B * T * C);
    float* x_cf    = alloc((size_t)B * C * T);
    float* q_cf    = alloc((size_t)B * C * T);
    float* dwbuf   = alloc((size_t)B * C * T);
    float* qb      = alloc((size_t)B * C * T);
    float* kb      = alloc((size_t)B * C * T);
    float* vb      = alloc((size_t)B * C * T);
    float* attno   = alloc((size_t)B * T * C);
    float* proj    = alloc((size_t)B * T * C);
    float* x_tok   = alloc((size_t)B * T * C);
    float* hln     = alloc((size_t)B * T * C);
    float* ffh     = alloc((size_t)B * T * 256);

    transpose_b_kernel<<<gblk((long long)B * C * T), 256, 0, stream>>>(feat, tokens, B, C, T); // -> [B,T,C]

    auto run_qkv = [&](const float* in_cf, int pbase, float* outb) {
        dwconv_kernel<<<gblk((long long)B * C * T), 256, 0, stream>>>(in_cf, P(pbase + 3), P(pbase + 2),
                                                                      dwbuf, B, C, T, T, 7, 1, 3);
        launch_gemm(stream, P(pbase + 5), dwbuf, outb, C, T, C, C, T, T,
                    0, (long long)C * T, (long long)C * T, B, 0,
                    P(pbase + 1), P(pbase + 4), P(pbase + 0),
                    nullptr, nullptr, nullptr, nullptr, 0, 2);
    };

    for (int l = 0; l < 3; ++l) {
        int Lb = 67 + l * 29;
        transpose_b_kernel<<<gblk((long long)B * T * C), 256, 0, stream>>>(tokens, x_cf, B, T, C); // -> [B,C,T]
        causal_conv_kernel<<<gblk((long long)B * C * T), 256, 0, stream>>>(x_cf, P(Lb + 1), P(Lb + 0), q_cf, B);
        run_qkv(q_cf, Lb + 16, qb);   // q dsconv
        run_qkv(x_cf, Lb + 6,  kb);   // k dsconv
        run_qkv(x_cf, Lb + 23, vb);   // v dsconv
        attention_kernel<<<B * 8, 128, 0, stream>>>(qb, kb, vb, P(Lb + 22), attno);
        // out projection [6400,64]@[64,64] + b
        launch_gemm(stream, attno, P(Lb + 15), proj, B * T, C, C, C, C, C,
                    0, 0, 0, 1, 0, nullptr, nullptr, nullptr,
                    nullptr, P(Lb + 14), nullptr, nullptr, 0, 0);
        residual1_kernel<<<gblk((long long)B * T * C), 256, 0, stream>>>(q_cf, proj, x_tok, B);
        layernorm_kernel<<<gblk(B * T), 256, 0, stream>>>(x_tok, P(Lb + 13), P(Lb + 12), hln, B * T, C);
        // FF1: relu(h@W1+b1)
        launch_gemm(stream, hln, P(Lb + 3), ffh, B * T, 256, C, C, 256, 256,
                    0, 0, 0, 1, 0, nullptr, nullptr, nullptr,
                    nullptr, P(Lb + 2), nullptr, nullptr, 0, 1);
        // FF2 + fused residual: tokens = x_tok + (ffh@W2+b2)
        launch_gemm(stream, ffh, P(Lb + 5), tokens, B * T, C, 256, 256, C, C,
                    0, 0, 0, 1, 0, nullptr, nullptr, nullptr,
                    nullptr, P(Lb + 4), nullptr, x_tok, 0, 0);
    }

    // ---------------- classifier ----------------
    float* lnf = alloc((size_t)B * T * C);
    layernorm_kernel<<<gblk(B * T), 256, 0, stream>>>(tokens, P(155), P(154), lnf, B * T, C);
    float* h512 = alloc((size_t)B * 512);
    // h = gelu(bn_g*(flat@W+b) + bn_b)
    launch_gemm(stream, lnf, P(2), h512, B, 512, T * C, T * C, 512, 512,
                0, 0, 0, 1, 0, nullptr, nullptr, nullptr,
                P(6), P(1), P(5), nullptr, 0, 2);
    // logits = h@W2 + b2
    launch_gemm(stream, h512, P(4), (float*)d_out, B, 5, 512, 512, 5, 5,
                0, 0, 0, 1, 0, nullptr, nullptr, nullptr,
                nullptr, P(3), nullptr, nullptr, 0, 0);
}
